// calOrientationGabor_39539468927747
// MI455X (gfx1250) — matrix operand order repaired and checked by reading, not compiled, not run on hardware
//
#include <hip/hip_runtime.h>
#include <hip/hip_bf16.h>
#include <math.h>

typedef __attribute__((ext_vector_type(16))) _Float16 v16h;
typedef __attribute__((ext_vector_type(8)))  float    v8f;

#define HH 768
#define WW 768
#define NPIX (HH*WW)
#define NUMK 180
#define NT 12              // N tiles of 16 -> 192 (pad)
#define KCH 11             // K chunks of 32 -> 352 (pad)
#define KROW 20            // padded patch-row length in K space (17 real + 3 zero)
#define PSTRIDE 88         // LDS patch row stride (halves)
#define PI_F 3.14159265358979323846f

#define BANK_HALVES (NT*KCH*32*16)      // 67584 halves = 135168 B
#define WS_MAXSLOT 0
#define WS_BANK    256
#define WS_VAR     (WS_BANK + BANK_HALVES*2)
#define WS_BEST    (WS_VAR + NPIX*4)

// ---------------------------------------------------------------------------
// Kernel 0: build Gabor bank directly in WMMA B-fragment layout (f16).
// B tile (kchunk x 16 orientations): lane = column n (lane&15), lane>>4 picks
// K half (0-15 / 16-31), v16h element e -> K = 16*(lane>>4)+e.  Also zero the
// global-max slot for this launch.
// ---------------------------------------------------------------------------
__global__ void gabor_bank_kernel(_Float16* __restrict__ bank,
                                  unsigned int* __restrict__ maxslot) {
  int idx = blockIdx.x * 256 + threadIdx.x;
  if (idx == 0) *maxslot = 0u;
  if (idx >= BANK_HALVES) return;
  int e  = idx & 15;
  int L  = (idx >> 4) & 31;
  int tl = idx >> 9;            // (t*KCH + kc)
  int kc = tl % KCH;
  int t  = tl / KCH;
  int n  = 16 * t + (L & 15);
  int g  = L >> 4;
  int k  = 32 * kc + 16 * g + e;
  int ky = k / KROW;
  int kx = k - ky * KROW;
  float val = 0.f;
  if (n < NUMK && ky < 17 && kx < 17) {
    float th = PI_F * (float)n / 180.f;
    float ct = cosf(th), st = sinf(th);
    float xi = (float)ky - 8.5f;   // c[i] = i - 8.5
    float yj = (float)kx - 8.5f;
    float xt =  xi * ct + yj * st;
    float yt = -xi * st + yj * ct;
    val = expf(-0.5f * (xt*xt / (1.8f*1.8f) + yt*yt / (2.4f*2.4f))) *
          cosf((2.0f * PI_F / 4.0f) * xt);
  }
  bank[idx] = (_Float16)val;
}

// ---------------------------------------------------------------------------
// Kernel 1: im2col GEMM conv via v_wmma_f32_16x16x32_f16 + in-register
// channel reductions.  Block = 128 threads (4 waves); each wave owns 16
// consecutive pixels of row y; workgroup covers 64 pixels.
// ---------------------------------------------------------------------------
__global__ __launch_bounds__(128) void gabor_conv_kernel(
    const float* __restrict__ img, const _Float16* __restrict__ bank,
    float* __restrict__ varmap, float* __restrict__ bestmap,
    unsigned int* __restrict__ maxslot) {
  __shared__ _Float16 patch[18 * PSTRIDE];   // rows 0..16 real, row 17 zero pad

  const int y   = blockIdx.y;
  const int x0  = blockIdx.x * 64;
  const int tid = threadIdx.x;

  // Stage f32 halo -> f16 LDS, zero-filling conv padding + layout padding.
  for (int d = tid; d < 18 * (PSTRIDE / 2); d += 128) {
    int row = d / (PSTRIDE / 2);
    int c0  = (d - row * (PSTRIDE / 2)) * 2;
    float f0 = 0.f, f1 = 0.f;
    int iy = y + row - 8;
    if (row < 17 && iy >= 0 && iy < HH) {
      int ix0 = x0 - 8 + c0;
      if (c0 < 80 && ix0 >= 0 && ix0 < WW)             f0 = img[iy * WW + ix0];
      if ((c0+1) < 80 && (ix0+1) >= 0 && (ix0+1) < WW) f1 = img[iy * WW + ix0 + 1];
    }
    patch[row * PSTRIDE + c0]     = (_Float16)f0;
    patch[row * PSTRIDE + c0 + 1] = (_Float16)f1;
  }
  __syncthreads();

  const int lane  = tid & 31;
  const int wv    = tid >> 5;          // wave id -> pixel sub-tile
  const int mrow  = lane & 15;         // A-matrix row M
  const int g     = lane >> 4;         // lane group (K half)
  const int xbase = 16 * wv + mrow;    // patch column base for this lane

  v8f acc[NT];
  #pragma unroll
  for (int t = 0; t < NT; ++t) acc[t] = v8f{};

  const v16h* __restrict__ bptr = (const v16h*)bank;

  for (int kc = 0; kc < KCH; ++kc) {
    // ---- batch-issue all 12 B fragments for this chunk (24 x b128) ----
    v16h b[NT];
    #pragma unroll
    for (int t = 0; t < NT; ++t) b[t] = bptr[(t * KCH + kc) * 32 + lane];

    // prefetch next chunk's first B tile while this chunk computes
    if (kc + 1 < KCH) __builtin_prefetch(&bptr[(kc + 1) * 32 + lane], 0, 1);

    // ---- assemble A fragment from LDS (contiguous K = 16*g + e, same order as B) ----
    const int kb = 32 * kc + 16 * g;
    v16h a;
    #pragma unroll
    for (int e = 0; e < 16; ++e) {
      const int ce = e;
      int k  = kb + ce;
      int ky = k / KROW;
      int kx = k - ky * KROW;
      a[e] = patch[ky * PSTRIDE + xbase + kx];
    }
    #pragma unroll
    for (int t = 0; t < NT; ++t) {
      acc[t] = __builtin_amdgcn_wmma_f32_16x16x32_f16(
          false, a, false, b[t], (short)0, acc[t], false, false);
    }
  }

  // ---- per-pixel reductions over 180 channels (C layout: lane = N col,
  //      vgpr v = M row v (lanes 0-15) / v+8 (lanes 16-31)) ----
  const int ncol = lane & 15;
  float wave_max = 0.f;
  #pragma unroll
  for (int v = 0; v < 8; ++v) {
    float r[NT];
    float mx = -1.f; int bn = 0;
    #pragma unroll
    for (int t = 0; t < NT; ++t) {
      int n = 16 * t + ncol;
      float rv = (n < NUMK) ? fabsf(acc[t][v]) : 0.f;
      r[t] = rv;
      if (n < NUMK && rv > mx) { mx = rv; bn = n; }
    }
    #pragma unroll
    for (int m = 1; m <= 8; m <<= 1) {
      float omx = __shfl_xor(mx, m, 32);
      int   obn = __shfl_xor(bn, m, 32);
      if (omx > mx || (omx == mx && obn < bn)) { mx = omx; bn = obn; }
    }
    // orient_diff(n) = (pi/180) * min(|bn-n|, 180-|bn-n|)  (exact rewrite)
    float s = 0.f;
    #pragma unroll
    for (int t = 0; t < NT; ++t) {
      int n = 16 * t + ncol;
      if (n < NUMK) {
        int   di  = abs(bn - n);
        int   odi = min(di, NUMK - di);
        float od  = (float)odi * (PI_F / 180.f);
        float rd  = r[t] - mx;
        s += od * rd * rd;
      }
    }
    #pragma unroll
    for (int m = 1; m <= 8; m <<= 1) s += __shfl_xor(s, m, 32);
    float var     = sqrtf(s);
    bool  upd     = var > 0.f;               // variance_data initialized to 0
    float vout    = upd ? var : 0.f;
    float bestang = upd ? ((float)bn * (PI_F / 180.f)) : 0.f;
    wave_max = fmaxf(wave_max, vout);
    if (ncol == v) {                         // lane v -> m=v, lane 16+v -> m=v+8
      int x = x0 + 16 * wv + v + 8 * g;
      int p = y * WW + x;
      varmap[p]  = vout;
      bestmap[p] = bestang;
    }
  }
  #pragma unroll
  for (int m = 1; m <= 16; m <<= 1)
    wave_max = fmaxf(wave_max, __shfl_xor(wave_max, m, 32));
  if (lane == 0) atomicMax(maxslot, __float_as_uint(wave_max));  // all >= 0
}

// ---------------------------------------------------------------------------
// Kernel 2: normalize variance by global max, confidence clip, sin/cos planes.
// d_out = [sin(best) | cos(best) | best | conf], each HxW f32.
// ---------------------------------------------------------------------------
__global__ void gabor_finalize_kernel(const float* __restrict__ varmap,
                                      const float* __restrict__ bestmap,
                                      const unsigned int* __restrict__ maxslot,
                                      float* __restrict__ out) {
  int p = blockIdx.x * 256 + threadIdx.x;
  if (p >= NPIX) return;
  float vmax = __uint_as_float(*maxslot);
  float var  = varmap[p];
  float best = bestmap[p];
  float vn   = var / vmax;                          // matches reference /max
  float conf = fminf(fmaxf(vn * (1.0f / 0.2f), 0.f), 1.f);
  // THRESHOLD == 0 and conf >= 0 -> where() is a no-op
  float sb, cb;
  sincosf(best, &sb, &cb);
  out[p]            = sb;
  out[NPIX + p]     = cb;
  out[2 * NPIX + p] = best;
  out[3 * NPIX + p] = conf;
}

extern "C" void kernel_launch(void* const* d_in, const int* in_sizes, int n_in,
                              void* d_out, int out_size, void* d_ws, size_t ws_size,
                              hipStream_t stream) {
  (void)in_sizes; (void)n_in; (void)out_size; (void)ws_size;
  const float* img = (const float*)d_in[0];   // image (1,1,768,768) f32
  // d_in[1] = label (unused by reference), d_in[2] = iter (== 1)

  char* ws = (char*)d_ws;
  unsigned int* maxslot = (unsigned int*)(ws + WS_MAXSLOT);
  _Float16*     bank    = (_Float16*)(ws + WS_BANK);
  float*        varmap  = (float*)(ws + WS_VAR);
  float*        bestmap = (float*)(ws + WS_BEST);
  float*        out     = (float*)d_out;

  gabor_bank_kernel<<<(BANK_HALVES + 255) / 256, 256, 0, stream>>>(bank, maxslot);
  dim3 grid(WW / 64, HH);
  gabor_conv_kernel<<<grid, 128, 0, stream>>>(img, bank, varmap, bestmap, maxslot);
  gabor_finalize_kernel<<<(NPIX + 255) / 256, 256, 0, stream>>>(varmap, bestmap, maxslot, out);
}
